// Net_8126078124451
// MI455X (gfx1250) — matrix operand order repaired and checked by reading, not compiled, not run on hardware
//
#include <hip/hip_runtime.h>
#include <hip/hip_bf16.h>

#define KK 5
#define KC 125
#define KCP 128   // layer-1 A row stride (padded so GEMM A rows are 16B aligned)

typedef __attribute__((ext_vector_type(16))) __bf16 v16bf;
typedef __attribute__((ext_vector_type(4)))  __bf16 v4bf;
typedef __attribute__((ext_vector_type(8)))  float  v8f;
typedef __attribute__((ext_vector_type(4)))  float  v4f;

// ---------------------------------------------------------------- zero
__global__ void zero_f(float* __restrict__ p, long long n) {
  long long i = (long long)blockIdx.x * blockDim.x + threadIdx.x;
  long long st = (long long)gridDim.x * blockDim.x;
  for (; i < n; i += st) p[i] = 0.0f;
}

// ---------------------------------------------------------------- WMMA GEMM
// C[M,N] = A[M,Kt] * B[Kt,N]; A row stride lda (>=Kt, multiple of 4).
// fp32 in memory, bf16 fragments, fp32 accumulate.
// Block = 256 thr = 8 wave32, waves tiled 2(M) x 4(N); block tile 32x64.
// K iterated 64 wide (two 32-sub-chunks per barrier pair, 2 WMMA per wave).
// Steady state takes a UNIFORM fast path: unguarded b128 A loads + b32 B
// loads, no exec masking.  Edge tiles use multiply-masked (still branchless)
// loads.
__global__ __launch_bounds__(256)
void wmma_gemm(const float* __restrict__ A, const float* __restrict__ B,
               float* __restrict__ C, int M, int N, int Kt, int lda) {
  __shared__ __align__(32) __bf16 ldsA[2][2 * 512];
  __shared__ __align__(32) __bf16 ldsB[2][4 * 512];
  const int tid  = threadIdx.x;
  const int wave = tid >> 5;
  const int lane = tid & 31;
  const int m0 = blockIdx.y * 32;
  const int n0 = blockIdx.x * 64;
  const int tA = wave >> 2;
  const int uB = wave & 3;
  v8f acc = {};

  // ---- per-thread staging coordinates (loop-invariant) ----
  // A: row mloc (0..31), 4-wide k segment ksegA
  const int  mloc  = tid >> 3;
  const int  ksegA = (tid & 7) << 2;            // 0,4,...,28
  const int  gm    = m0 + mloc;
  const bool mok   = gm < M;
  const float* Arow = A + (size_t)(mok ? gm : (M - 1)) * lda;
  // A frag layout: lane L (m=L%16, half=L/16), elem i: k=(i/8)*16+half*8+(i%8)
  const int offA = (mloc >> 4) * 512 +
                   ((((ksegA & 15) >> 3) * 16 + (mloc & 15)) * 16) +
                   ((ksegA >> 4) * 8) + (ksegA & 7);
  // B: column nl (same for both tasks), two 4-wide k segments 16 apart
  const int  nl    = tid & 63;
  const int  ksegB = ((tid >> 6) << 2);         // 0,4,8,12 ; second task +16
  const int  gn    = n0 + nl;
  const bool nok   = gn < N;
  const int  gnc   = nok ? gn : (N - 1);
  // B frag layout: lane L (n=L%16, half=L/16), elem i: k=(i/8)*16+half*8+(i%8)
  const int offB0 = (nl >> 4) * 512 + ((((ksegB >> 3) & 1) * 16 + (nl & 15)) * 16) + (ksegB & 7);
  const int offB1 = (nl >> 4) * 512 + ((((ksegB >> 3) & 1) * 16 + (nl & 15)) * 16) + 8 + (ksegB & 7);

  const bool fullMN = (m0 + 32 <= M) && (n0 + 64 <= N);

  const int nchunks = (Kt + 63) >> 6;
  for (int ch = 0; ch < nchunks; ++ch) {
    if (ch + 1 < nchunks) {                    // gfx1250 global_prefetch_b8
      int pk = ((ch + 1) << 6) + ksegA;
      if (pk >= Kt) pk = Kt - 1;
      __builtin_prefetch(&Arow[pk], 0, 3);
    }
    #pragma unroll
    for (int sub = 0; sub < 2; ++sub) {
      const int k0 = (ch << 6) + (sub << 5);
      v4bf pa, pb0, pb1;
      if (fullMN && (k0 + 32 <= Kt)) {
        // ---------- uniform fast path: raw loads ----------
        v4f av = *reinterpret_cast<const v4f*>(Arow + k0 + ksegA);
        #pragma unroll
        for (int i = 0; i < 4; ++i) pa[i] = (__bf16)av[i];
        #pragma unroll
        for (int i = 0; i < 4; ++i)
          pb0[i] = (__bf16)B[(size_t)(k0 + ksegB + i) * N + gn];
        #pragma unroll
        for (int i = 0; i < 4; ++i)
          pb1[i] = (__bf16)B[(size_t)(k0 + ksegB + 16 + i) * N + gn];
      } else {
        // ---------- edge path: multiply-masked, still branchless ----------
        #pragma unroll
        for (int i = 0; i < 4; ++i) {
          int gk  = k0 + ksegA + i;
          int gkc = gk < Kt ? gk : Kt - 1;
          float v = Arow[gkc];
          pa[i] = (__bf16)(v * ((mok & (gk < Kt)) ? 1.0f : 0.0f));
        }
        #pragma unroll
        for (int i = 0; i < 4; ++i) {
          int gk  = k0 + ksegB + i;
          int gkc = gk < Kt ? gk : Kt - 1;
          float v = B[(size_t)gkc * N + gnc];
          pb0[i] = (__bf16)(v * ((nok & (gk < Kt)) ? 1.0f : 0.0f));
        }
        #pragma unroll
        for (int i = 0; i < 4; ++i) {
          int gk  = k0 + ksegB + 16 + i;
          int gkc = gk < Kt ? gk : Kt - 1;
          float v = B[(size_t)gkc * N + gnc];
          pb1[i] = (__bf16)(v * ((nok & (gk < Kt)) ? 1.0f : 0.0f));
        }
      }
      *reinterpret_cast<v4bf*>(&ldsA[sub][offA])  = pa;
      *reinterpret_cast<v4bf*>(&ldsB[sub][offB0]) = pb0;
      *reinterpret_cast<v4bf*>(&ldsB[sub][offB1]) = pb1;
    }
    __syncthreads();
    #pragma unroll
    for (int sub = 0; sub < 2; ++sub) {
      v16bf af = *reinterpret_cast<const v16bf*>(&ldsA[sub][tA * 512 + lane * 16]);
      v16bf bf = *reinterpret_cast<const v16bf*>(&ldsB[sub][uB * 512 + lane * 16]);
      acc = __builtin_amdgcn_wmma_f32_16x16x32_bf16(false, af, false, bf,
                                                    (short)0, acc, false, false);
    }
    __syncthreads();
  }
  // C/D 16x16 layout: lane L (n=L%16, half=L/16), VGPR r -> M = r + half*8
  const int half  = lane >> 4;
  const int nn    = lane & 15;
  const int crow0 = m0 + tA * 16 + half * 8;
  const int ccol  = n0 + uB * 16 + nn;
  if (fullMN) {
    #pragma unroll
    for (int r = 0; r < 8; ++r)
      C[(size_t)(crow0 + r) * N + ccol] = acc[r];
  } else if (ccol < N) {
    #pragma unroll
    for (int r = 0; r < 8; ++r) {
      int row = crow0 + r;
      if (row < M) C[(size_t)row * N + ccol] = acc[r];
    }
  }
}

// ---------------------------------------------------------------- spline scatter
__device__ __forceinline__ void spline_geom(const float* __restrict__ pos,
                                            int s, int t, float inv2r,
                                            float fr[3], int bi[3]) {
  #pragma unroll
  for (int d = 0; d < 3; ++d) {
    float u = (pos[t * 3 + d] - pos[s * 3 + d]) * inv2r + 0.5f;
    u = fminf(fmaxf(u, 0.0f), 1.0f);
    float v = u * (float)(KK - 1);
    float b = floorf(v);
    fr[d] = v - b;
    bi[d] = (int)b;
  }
}

__global__ void scatter_c1_k(const float* __restrict__ pos, const int* __restrict__ src,
                             const int* __restrict__ tgt, int E, float inv2r,
                             float* __restrict__ A, float* __restrict__ deg) {
  int e = blockIdx.x * blockDim.x + threadIdx.x;
  if (e >= E) return;
  int s = src[e], t = tgt[e];
  float fr[3]; int bi[3];
  spline_geom(pos, s, t, inv2r, fr, bi);
  #pragma unroll
  for (int q = 0; q < 8; ++q) {
    int bx = q & 1, by = (q >> 1) & 1, bz = (q >> 2) & 1;
    float w = (bx ? fr[0] : 1.0f - fr[0]) * (by ? fr[1] : 1.0f - fr[1]) *
              (bz ? fr[2] : 1.0f - fr[2]);
    int ix = min(bi[0] + bx, KK - 1), iy = min(bi[1] + by, KK - 1),
        iz = min(bi[2] + bz, KK - 1);
    int wi = ix + KK * iy + KK * KK * iz;
    atomicAdd(&A[(size_t)t * KCP + wi], w);   // x == ones; padded row stride
  }
  atomicAdd(&deg[t], 1.0f);
}

// 4 channels per thread
__global__ void scatter_c64_k(const float* __restrict__ pos, const int* __restrict__ src,
                              const int* __restrict__ tgt, int E, float inv2r,
                              const float* __restrict__ x,
                              float* __restrict__ A, float* __restrict__ deg) {
  long long gid = (long long)blockIdx.x * blockDim.x + threadIdx.x;
  if (gid >= (long long)E * 16) return;
  int e  = (int)(gid >> 4);
  int c0 = (int)(gid & 15) << 2;
  int s = src[e], t = tgt[e];
  float fr[3]; int bi[3];
  spline_geom(pos, s, t, inv2r, fr, bi);
  float xj[4];
  #pragma unroll
  for (int i = 0; i < 4; ++i) xj[i] = x[(size_t)s * 64 + c0 + i];
  #pragma unroll
  for (int q = 0; q < 8; ++q) {
    int bx = q & 1, by = (q >> 1) & 1, bz = (q >> 2) & 1;
    float w = (bx ? fr[0] : 1.0f - fr[0]) * (by ? fr[1] : 1.0f - fr[1]) *
              (bz ? fr[2] : 1.0f - fr[2]);
    int ix = min(bi[0] + bx, KK - 1), iy = min(bi[1] + by, KK - 1),
        iz = min(bi[2] + bz, KK - 1);
    int wi = ix + KK * iy + KK * KK * iz;
    float* base = &A[((size_t)t * KC + wi) * 64 + c0];
    #pragma unroll
    for (int i = 0; i < 4; ++i) atomicAdd(base + i, w * xj[i]);
  }
  if ((gid & 15) == 0) atomicAdd(&deg[t], 1.0f);
}

// ---------------------------------------------------------------- elementwise
__device__ __forceinline__ float elu1(float v) { return v > 0.0f ? v : (expf(v) - 1.0f); }

__global__ void fuse1_k(const float* __restrict__ msg, const float* __restrict__ deg,
                        const float* __restrict__ R1, const float* __restrict__ b1,
                        float* __restrict__ out, int n) {
  long long gid = (long long)blockIdx.x * blockDim.x + threadIdx.x;
  if (gid >= (long long)n * 64) return;
  int i = (int)(gid >> 6), c = (int)(gid & 63);
  out[gid] = elu1(msg[gid] / fmaxf(deg[i], 1.0f) + R1[c] + b1[c]);
}

__global__ void fuse2_k(const float* __restrict__ msg, const float* __restrict__ deg,
                        const float* __restrict__ xR, const float* __restrict__ b,
                        float* __restrict__ out, int n, int C) {
  long long gid = (long long)blockIdx.x * blockDim.x + threadIdx.x;
  if (gid >= (long long)n * C) return;
  int i = (int)(gid / C), c = (int)(gid % C);
  out[gid] = elu1(msg[gid] / fmaxf(deg[i], 1.0f) + xR[gid] + b[c]);
}

__global__ void gather_rows_k(const float* __restrict__ src, const int* __restrict__ idx,
                              float* __restrict__ dst, int n, int C) {
  long long gid = (long long)blockIdx.x * blockDim.x + threadIdx.x;
  if (gid >= (long long)n * C) return;
  int j = (int)(gid / C), c = (int)(gid % C);
  dst[gid] = src[(size_t)idx[j] * C + c];
}

__global__ void pool_k(const float* __restrict__ x3, const int* __restrict__ batch,
                       const int* __restrict__ idx1, const int* __restrict__ idx2,
                       float* __restrict__ g, float* __restrict__ cnt, int n3) {
  long long gid = (long long)blockIdx.x * blockDim.x + threadIdx.x;
  if (gid >= (long long)n3 * 128) return;
  int k = (int)(gid >> 7), c = (int)(gid & 127);
  int b = batch[idx1[idx2[k]]];
  atomicAdd(&g[b * 128 + c], x3[gid]);
  if (c == 0) atomicAdd(&cnt[b], 1.0f);
}

__global__ void mean_k(float* __restrict__ g, const float* __restrict__ cnt) {
  int gid = blockIdx.x * blockDim.x + threadIdx.x;
  if (gid >= 16 * 128) return;
  g[gid] /= fmaxf(cnt[gid >> 7], 1.0f);
}

__global__ void bias_elu_k(float* __restrict__ h, const float* __restrict__ b, int n, int C) {
  int gid = blockIdx.x * blockDim.x + threadIdx.x;
  if (gid >= n * C) return;
  h[gid] = elu1(h[gid] + b[gid % C]);
}

__global__ void logsoftmax_k(const float* __restrict__ logits,
                             const float* __restrict__ lb, float* __restrict__ out) {
  int t = threadIdx.x;
  if (t >= 16) return;
  float z[10];
  float m = -1e30f;
  #pragma unroll
  for (int c = 0; c < 10; ++c) { z[c] = logits[t * 10 + c] + lb[c]; m = fmaxf(m, z[c]); }
  float s = 0.0f;
  #pragma unroll
  for (int c = 0; c < 10; ++c) s += expf(z[c] - m);
  float lse = m + logf(s);
  #pragma unroll
  for (int c = 0; c < 10; ++c) out[t * 10 + c] = z[c] - lse;
}

// ---------------------------------------------------------------- launch
static inline unsigned nblk(long long n, int bs) { return (unsigned)((n + bs - 1) / bs); }

extern "C" void kernel_launch(void* const* d_in, const int* in_sizes, int n_in,
                              void* d_out, int out_size, void* d_ws, size_t ws_size,
                              hipStream_t stream) {
  const float* pos  = (const float*)d_in[0];
  const int*   batch= (const int*)d_in[1];
  const int* src1 = (const int*)d_in[2]; const int* tgt1 = (const int*)d_in[3];
  const int* src2 = (const int*)d_in[4]; const int* tgt2 = (const int*)d_in[5];
  const int* src3 = (const int*)d_in[6]; const int* tgt3 = (const int*)d_in[7];
  const int* idx1 = (const int*)d_in[8]; const int* idx2 = (const int*)d_in[9];
  const float* W1 = (const float*)d_in[10]; const float* R1 = (const float*)d_in[11]; const float* b1 = (const float*)d_in[12];
  const float* W2 = (const float*)d_in[13]; const float* R2 = (const float*)d_in[14]; const float* b2 = (const float*)d_in[15];
  const float* W3 = (const float*)d_in[16]; const float* R3 = (const float*)d_in[17]; const float* b3 = (const float*)d_in[18];
  const float* lw1 = (const float*)d_in[19]; const float* lb1 = (const float*)d_in[20];
  const float* lw2 = (const float*)d_in[21]; const float* lb2 = (const float*)d_in[22];
  const float* lw3 = (const float*)d_in[23]; const float* lb3 = (const float*)d_in[24];

  const int n1 = in_sizes[0] / 3;
  const int E1 = in_sizes[2], E2 = in_sizes[4], E3 = in_sizes[6];
  const int n2 = in_sizes[8], n3 = in_sizes[9];

  float* ws = (float*)d_ws;
  size_t cur = 0;
  auto alloc = [&](size_t nf) -> float* {
    float* p = ws + cur;
    cur += (nf + 63) & ~(size_t)63;
    return p;
  };
  size_t amax = (size_t)n1 * KCP;
  if ((size_t)n2 * KC * 64 > amax) amax = (size_t)n2 * KC * 64;
  if ((size_t)n3 * KC * 64 > amax) amax = (size_t)n3 * KC * 64;
  size_t xrmax = (size_t)n2 * 64;
  if ((size_t)n3 * 128 > xrmax) xrmax = (size_t)n3 * 128;

  float* A     = alloc(amax);                 // shared A buffer, all layers
  float* deg   = alloc((size_t)n1);
  float* msg   = alloc((size_t)n1 * 64);
  float* xR    = alloc(xrmax);
  float* x1    = alloc((size_t)n1 * 64);
  float* x2in  = alloc((size_t)n2 * 64);
  float* x2out = alloc((size_t)n2 * 64);
  float* x3in  = alloc((size_t)n3 * 64);
  float* x3out = alloc((size_t)n3 * 128);
  float* pos2  = alloc((size_t)n2 * 3);
  float* pos3  = alloc((size_t)n3 * 3);
  float* gpool = alloc(16 * 128);
  float* cnt   = alloc(64);
  float* h1    = alloc(16 * 256);
  float* h2    = alloc(16 * 256);
  float* logit = alloc(16 * 16);

  dim3 blk(256);

  // ---- layer 1: Cin=1, Cout=64, r=0.2 (A padded to row stride 128)
  { unsigned zb = nblk((long long)n1 * KCP, 256); if (zb > 32768u) zb = 32768u;
    zero_f<<<zb, blk, 0, stream>>>(A, (long long)n1 * KCP); }
  zero_f<<<nblk(n1, 256), blk, 0, stream>>>(deg, n1);
  scatter_c1_k<<<nblk(E1, 256), blk, 0, stream>>>(pos, src1, tgt1, E1, 1.0f / 0.4f, A, deg);
  { dim3 g(1, (unsigned)((n1 + 31) / 32));
    wmma_gemm<<<g, blk, 0, stream>>>(A, W1, msg, n1, 64, KC, KCP); }
  fuse1_k<<<nblk((long long)n1 * 64, 256), blk, 0, stream>>>(msg, deg, R1, b1, x1, n1);

  // ---- pool to level 2
  gather_rows_k<<<nblk((long long)n2 * 64, 256), blk, 0, stream>>>(x1, idx1, x2in, n2, 64);
  gather_rows_k<<<nblk((long long)n2 * 3, 256), blk, 0, stream>>>(pos, idx1, pos2, n2, 3);

  // ---- layer 2: Cin=64, Cout=64, r=0.4
  zero_f<<<32768u, blk, 0, stream>>>(A, (long long)n2 * KC * 64);
  zero_f<<<nblk(n2, 256), blk, 0, stream>>>(deg, n2);
  scatter_c64_k<<<nblk((long long)E2 * 16, 256), blk, 0, stream>>>(pos2, src2, tgt2, E2, 1.0f / 0.8f, x2in, A, deg);
  { dim3 g(1, (unsigned)((n2 + 31) / 32));
    wmma_gemm<<<g, blk, 0, stream>>>(A, W2, msg, n2, 64, KC * 64, KC * 64); }
  { dim3 g(1, (unsigned)((n2 + 31) / 32));
    wmma_gemm<<<g, blk, 0, stream>>>(x2in, R2, xR, n2, 64, 64, 64); }
  fuse2_k<<<nblk((long long)n2 * 64, 256), blk, 0, stream>>>(msg, deg, xR, b2, x2out, n2, 64);

  // ---- pool to level 3
  gather_rows_k<<<nblk((long long)n3 * 64, 256), blk, 0, stream>>>(x2out, idx2, x3in, n3, 64);
  gather_rows_k<<<nblk((long long)n3 * 3, 256), blk, 0, stream>>>(pos2, idx2, pos3, n3, 3);

  // ---- layer 3: Cin=64, Cout=128, r=1.0
  zero_f<<<32768u, blk, 0, stream>>>(A, (long long)n3 * KC * 64);
  zero_f<<<nblk(n3, 256), blk, 0, stream>>>(deg, n3);
  scatter_c64_k<<<nblk((long long)E3 * 16, 256), blk, 0, stream>>>(pos3, src3, tgt3, E3, 1.0f / 2.0f, x3in, A, deg);
  { dim3 g(2, (unsigned)((n3 + 31) / 32));
    wmma_gemm<<<g, blk, 0, stream>>>(A, W3, msg, n3, 128, KC * 64, KC * 64); }
  { dim3 g(2, (unsigned)((n3 + 31) / 32));
    wmma_gemm<<<g, blk, 0, stream>>>(x3in, R3, xR, n3, 128, 64, 64); }
  fuse2_k<<<nblk((long long)n3 * 128, 256), blk, 0, stream>>>(msg, deg, xR, b3, x3out, n3, 128);

  // ---- global mean pool
  zero_f<<<8u, blk, 0, stream>>>(gpool, 16 * 128);
  zero_f<<<1u, blk, 0, stream>>>(cnt, 64);
  pool_k<<<nblk((long long)n3 * 128, 256), blk, 0, stream>>>(x3out, batch, idx1, idx2, gpool, cnt, n3);
  mean_k<<<8u, blk, 0, stream>>>(gpool, cnt);

  // ---- MLP head (WMMA for the GEMMs)
  { dim3 g(4, 1); wmma_gemm<<<g, blk, 0, stream>>>(gpool, lw1, h1, 16, 256, 128, 128); }
  bias_elu_k<<<16u, blk, 0, stream>>>(h1, lb1, 16, 256);
  { dim3 g(4, 1); wmma_gemm<<<g, blk, 0, stream>>>(h1, lw2, h2, 16, 256, 256, 256); }
  bias_elu_k<<<16u, blk, 0, stream>>>(h2, lb2, 16, 256);
  { dim3 g(1, 1); wmma_gemm<<<g, blk, 0, stream>>>(h2, lw3, logit, 16, 10, 256, 256); }
  logsoftmax_k<<<1u, dim3(32), 0, stream>>>(logit, lb3, (float*)d_out);
}
